// MultiHeadAttention_22677427322920
// MI455X (gfx1250) — matrix operand order repaired and checked by reading, compiled, not run on hardware
//
#include <hip/hip_runtime.h>
#include <hip/hip_bf16.h>

// ---------------------------------------------------------------------------
// MultiHeadAttention forward for MI455X (gfx1250)
// bf16 WMMA GEMMs + fused flash attention + async global->LDS tile staging
// ---------------------------------------------------------------------------

typedef __attribute__((ext_vector_type(8)))  float   v8f;
typedef __attribute__((ext_vector_type(8)))  __bf16  bf16x8;
typedef __attribute__((ext_vector_type(16))) __bf16  bf16x16;

#define B_DIM   2
#define S_LEN   2048
#define DMODEL  1024
#define HEADS   16
#define DEPTH   64
#define NTOK    (B_DIM * S_LEN)   /* 4096 */
#define LDSS    72                /* padded LDS row stride (bf16 elems), 144B */

// ---- gfx1250 async global->LDS copy (ASYNCcnt-tracked, ISA §15.18 op 98) --
// LDS target address comes from the VDST VGPR; generic-pointer low 32 bits are
// the in-wave LDS byte offset (ISA §10.2 aperture mapping).
__device__ __forceinline__ void async_b128(const __bf16* g, const __bf16* l) {
  unsigned lds = (unsigned)(size_t)l;
  asm volatile("global_load_async_to_lds_b128 %0, %1, off"
               :: "v"(lds), "v"(g) : "memory");
}
__device__ __forceinline__ void wait_async_le(void) {   // older group done
  asm volatile("s_wait_asynccnt 0x8" ::: "memory");
}
__device__ __forceinline__ void wait_async_0(void) {    // everything done
  asm volatile("s_wait_asynccnt 0x0" ::: "memory");
}

// ---- WMMA fragment loaders (layouts per CDNA5 ISA §7.12.2) -----
// A-matrix 16x32 bf16: lane m (m=lane&15); VGPR0-3 hold K {0..7} (lanes 0-15)
// or K {8..15} (lanes 16-31); VGPR4-7 hold K {16..23} / {24..31}.
__device__ __forceinline__ bf16x16 fragA_ld(const __bf16* tile, int kbase, int lane) {
  const int row = lane & 15;
  const int off = kbase + ((lane >> 4) & 1) * 8;
  const __bf16* p = tile + row * LDSS + off;
  bf16x8 lo = *(const bf16x8*)p;          // K +0..7   (16B, aligned)
  bf16x8 hi = *(const bf16x8*)(p + 16);   // K +16..23 (16B, aligned)
  return __builtin_shufflevector(lo, hi, 0,1,2,3,4,5,6,7,8,9,10,11,12,13,14,15);
}

// B-matrix 32x16 bf16 (column-major operand): lane n (n=lane&15) holds column
// n, K {0..7,16..23} for lanes 0-15 and K {8..15,24..31} for lanes 16-31.
// `tile` rows here are the *columns* of the K x N matrix.
__device__ __forceinline__ bf16x16 fragB_ld(const __bf16* tile, int kbase, int lane) {
  const int col = lane & 15;
  const int off = kbase + ((lane >> 4) & 1) * 8;
  const __bf16* p = tile + col * LDSS + off;
  bf16x8 lo = *(const bf16x8*)p;          // K +0..7
  bf16x8 hi = *(const bf16x8*)(p + 16);   // K +16..23
  return __builtin_shufflevector(lo, hi, 0,1,2,3,4,5,6,7,8,9,10,11,12,13,14,15);
}

__device__ __forceinline__ v8f wmma_bf16(bf16x16 a, bf16x16 b, v8f c) {
  return __builtin_amdgcn_wmma_f32_16x16x32_bf16(
      /*neg_a=*/false, a, /*neg_b=*/false, b,
      /*c_mod=*/(short)0, c, /*reuse_a=*/false, /*reuse_b=*/false);
}

// ---------------------------------------------------------------------------
// f32 -> bf16 conversion (8 elements / thread, RNE)
// ---------------------------------------------------------------------------
__global__ __launch_bounds__(256) void cvt_bf16_kernel(
    const float* __restrict__ src, __bf16* __restrict__ dst, long n) {
  long i = (((long)blockIdx.x * blockDim.x) + threadIdx.x) * 8;
  if (i >= n) return;
  float4 a = *(const float4*)(src + i);
  float4 b = *(const float4*)(src + i + 4);
  bf16x8 o;
  o[0] = (__bf16)a.x; o[1] = (__bf16)a.y; o[2] = (__bf16)a.z; o[3] = (__bf16)a.w;
  o[4] = (__bf16)b.x; o[5] = (__bf16)b.y; o[6] = (__bf16)b.z; o[7] = (__bf16)b.w;
  *(bf16x8*)(dst + i) = o;
}

// ---------------------------------------------------------------------------
// C[M,N] = A[M,K] @ B[N,K]^T + bias   (bf16 in, f32 accumulate)
// 64x64 tile per 128-thread (4-wave) workgroup. K staged in 64-deep chunks
// via DOUBLE-BUFFERED async global->LDS copies: chunk i+1 is in flight while
// chunk i feeds the WMMAs; s_wait_asynccnt 8 retires only the older group.
// ---------------------------------------------------------------------------
template <bool OUT_BF16>
__global__ __launch_bounds__(128) void gemm_bt_kernel(
    const __bf16* __restrict__ A, const __bf16* __restrict__ B,
    const float* __restrict__ bias, void* __restrict__ Cv,
    int M, int N, int K) {
  __shared__ __align__(16) __bf16 As[2][64][LDSS];
  __shared__ __align__(16) __bf16 Bs[2][64][LDSS];

  const int tid = threadIdx.x, lane = tid & 31, wave = tid >> 5;
  const int n0 = blockIdx.x * 64, m0 = blockIdx.y * 64;

  // 8 async b128 ops per thread per chunk (4 for A-tile, 4 for B-tile)
  auto issue_chunk = [&](int buf, int k0) {
    for (int c = tid; c < 512; c += 128) {
      const int r = c >> 3, k8 = (c & 7) * 8;
      async_b128(A + (size_t)(m0 + r) * K + k0 + k8, &As[buf][r][k8]);
      async_b128(B + (size_t)(n0 + r) * K + k0 + k8, &Bs[buf][r][k8]);
    }
  };

  const v8f vzero = {};
  v8f acc[4] = {vzero, vzero, vzero, vzero};

  const int nchunk = K / 64;
  issue_chunk(0, 0);
  for (int i = 0; i < nchunk; i++) {
    const int buf = i & 1;
    if (i + 1 < nchunk) {
      issue_chunk(buf ^ 1, (i + 1) * 64);  // prefetch next chunk
      wait_async_le();                     // retire this chunk's 8 ops only
    } else {
      wait_async_0();
    }
    __syncthreads();                       // chunk i visible to all waves

#pragma unroll
    for (int ks = 0; ks < 2; ks++) {
      // preload all fragments of this k-step, then issue the 4 WMMAs
      bf16x16 a = fragA_ld(&As[buf][wave * 16][0], ks * 32, lane);
      bf16x16 bfr[4];
#pragma unroll
      for (int nt = 0; nt < 4; nt++)
        bfr[nt] = fragB_ld(&Bs[buf][nt * 16][0], ks * 32, lane);
#pragma unroll
      for (int nt = 0; nt < 4; nt++)
        acc[nt] = wmma_bf16(a, bfr[nt], acc[nt]);
    }
    __syncthreads();                       // reads done before buf is reused
  }

  // Epilogue: C/D layout = VGPR r: row (half*8 + r), col = lane&15 (+16*nt)
  const int ln = lane & 15, half = (lane >> 4) & 1;
#pragma unroll
  for (int nt = 0; nt < 4; nt++) {
#pragma unroll
    for (int r = 0; r < 8; r++) {
      const int row = m0 + wave * 16 + half * 8 + r;
      const int col = n0 + nt * 16 + ln;
      const float o = acc[nt][r] + bias[col];
      if (OUT_BF16) ((__bf16*)Cv)[(size_t)row * N + col] = (__bf16)o;
      else          ((float*)Cv)[(size_t)row * N + col]  = o;
    }
  }
}

// ---------------------------------------------------------------------------
// Flash attention: one block per (b, h, 64-query tile). 4 waves x 16 rows.
// logits = (Q Kh^T)/8 + mask*(-1e9); online softmax; O = P V.
// Q and K tiles staged with async global->LDS; V goes through VGPRs because
// it is transposed on the way into LDS.
// ---------------------------------------------------------------------------
__global__ __launch_bounds__(128) void flash_attn_kernel(
    const __bf16* __restrict__ Qp, const __bf16* __restrict__ Kp,
    const __bf16* __restrict__ Vp, const float* __restrict__ mask,
    __bf16* __restrict__ Out) {
  const int tid = threadIdx.x, lane = tid & 31, wave = tid >> 5;
  const int ln = lane & 15, half = (lane >> 4) & 1;
  const int q0 = blockIdx.x * 64;   // query tile
  const int h  = blockIdx.y;        // head
  const int b  = blockIdx.z;        // batch

  const __bf16* Qb = Qp + ((size_t)b * S_LEN) * DMODEL + h * DEPTH;
  const __bf16* Kb = Kp + ((size_t)b * S_LEN) * DMODEL + h * DEPTH;
  const __bf16* Vb = Vp + ((size_t)b * S_LEN) * DMODEL + h * DEPTH;
  const float*  mrow = mask + (size_t)b * S_LEN;

  __shared__ __align__(16) __bf16 Qs[64][LDSS];       // [query][depth]
  __shared__ __align__(16) __bf16 Ks[64][LDSS];       // [key][depth]
  __shared__ __align__(16) __bf16 Vts[64][LDSS];      // transposed: [depth][key]
  __shared__ __align__(16) __bf16 Ps[4][16][LDSS];    // per-wave P staging

  // Q tile (rows q0..q0+63, depth slice of this head): async, waited in iter 0
  for (int c = tid; c < 512; c += 128) {
    const int r = c >> 3, k8 = (c & 7) * 8;
    async_b128(Qb + (size_t)(q0 + r) * DMODEL + k8, &Qs[r][k8]);
  }

  const v8f vzero = {};
  v8f acc[4] = {vzero, vzero, vzero, vzero};
  float m_r[8], l_r[8];
#pragma unroll
  for (int r = 0; r < 8; r++) { m_r[r] = -1e30f; l_r[r] = 0.0f; }

  for (int kv = 0; kv < S_LEN; kv += 64) {
    __syncthreads();                       // previous tile's reads complete
    // K tile row-major: async straight into LDS
    for (int c = tid; c < 512; c += 128) {
      const int r = c >> 3, k8 = (c & 7) * 8;
      async_b128(Kb + (size_t)(kv + r) * DMODEL + k8, &Ks[r][k8]);
    }
    // V tile transposed into LDS (coalesced global read, scattered DS write)
    for (int c = tid; c < 512; c += 128) {
      const int r = c >> 3, k8 = (c & 7) * 8;
      bf16x8 vv = *(const bf16x8*)(Vb + (size_t)(kv + r) * DMODEL + k8);
#pragma unroll
      for (int j = 0; j < 8; j++) Vts[k8 + j][r] = vv[j];
    }
    wait_async_0();                        // K (and first-iter Q) landed
    __syncthreads();

    // additive mask term, one value per N-column (key) this lane owns
    float maskv[4];
#pragma unroll
    for (int nt = 0; nt < 4; nt++) maskv[nt] = -1e9f * mrow[kv + nt * 16 + ln];

    // S = Q K^T (depth=64 -> 2 WMMA k-steps x 4 key tiles)
    v8f s[4] = {vzero, vzero, vzero, vzero};
#pragma unroll
    for (int ks = 0; ks < 2; ks++) {
      bf16x16 a = fragA_ld(&Qs[wave * 16][0], ks * 32, lane);
      bf16x16 bfr[4];
#pragma unroll
      for (int nt = 0; nt < 4; nt++)
        bfr[nt] = fragB_ld(&Ks[nt * 16][0], ks * 32, lane);
#pragma unroll
      for (int nt = 0; nt < 4; nt++)
        s[nt] = wmma_bf16(a, bfr[nt], s[nt]);
    }
#pragma unroll
    for (int nt = 0; nt < 4; nt++)
#pragma unroll
      for (int r = 0; r < 8; r++)
        s[nt][r] = s[nt][r] * 0.125f + maskv[nt];   // 1/sqrt(64) then mask

    // Online softmax, per row r (row = half*8 + r); reduce across 16 lanes.
#pragma unroll
    for (int r = 0; r < 8; r++) {
      float mx = fmaxf(fmaxf(s[0][r], s[1][r]), fmaxf(s[2][r], s[3][r]));
      mx = fmaxf(mx, __shfl_xor(mx, 1, 16));
      mx = fmaxf(mx, __shfl_xor(mx, 2, 16));
      mx = fmaxf(mx, __shfl_xor(mx, 4, 16));
      mx = fmaxf(mx, __shfl_xor(mx, 8, 16));
      const float m_new = fmaxf(m_r[r], mx);
      const float corr  = __expf(m_r[r] - m_new);
      float psum = 0.0f;
#pragma unroll
      for (int nt = 0; nt < 4; nt++) {
        const float p = __expf(s[nt][r] - m_new);
        s[nt][r] = p;
        psum += p;
      }
      psum += __shfl_xor(psum, 1, 16);
      psum += __shfl_xor(psum, 2, 16);
      psum += __shfl_xor(psum, 4, 16);
      psum += __shfl_xor(psum, 8, 16);
      l_r[r] = l_r[r] * corr + psum;
      m_r[r] = m_new;
#pragma unroll
      for (int nt = 0; nt < 4; nt++) acc[nt][r] *= corr;   // rescale O
    }

    // Stage P (C-layout -> row-major LDS) so it can be re-read in A-layout.
    // Wave-private region; same-wave DS ops are in-order, no barrier needed.
#pragma unroll
    for (int nt = 0; nt < 4; nt++)
#pragma unroll
      for (int r = 0; r < 8; r++)
        Ps[wave][half * 8 + r][nt * 16 + ln] = (__bf16)s[nt][r];

    // O += P V  (keys=64 -> 2 WMMA k-steps x 4 depth tiles)
#pragma unroll
    for (int ks = 0; ks < 2; ks++) {
      bf16x16 a = fragA_ld(&Ps[wave][0][0], ks * 32, lane);
      bf16x16 bfr[4];
#pragma unroll
      for (int nt = 0; nt < 4; nt++)
        bfr[nt] = fragB_ld(&Vts[nt * 16][0], ks * 32, lane);
#pragma unroll
      for (int nt = 0; nt < 4; nt++)
        acc[nt] = wmma_bf16(a, bfr[nt], acc[nt]);
    }
  }

  // Normalize and store into concat layout [B, S, H*DEPTH] (bf16 for Wo GEMM)
#pragma unroll
  for (int nt = 0; nt < 4; nt++)
#pragma unroll
    for (int r = 0; r < 8; r++) {
      const int row = q0 + wave * 16 + half * 8 + r;
      const int col = h * DEPTH + nt * 16 + ln;
      const float o = acc[nt][r] / l_r[r];
      Out[((size_t)b * S_LEN + row) * DMODEL + col] = (__bf16)o;
    }
}

// ---------------------------------------------------------------------------
// Host-side orchestration. Workspace layout (256B aligned, 64 MiB total):
//  q/k/v bf16 (3x8MiB), Wq/Wk/Wv/Wo bf16 (4x2MiB), Qp/Kp/Vp bf16 (3x8MiB),
//  attn bf16 (8MiB).
// ---------------------------------------------------------------------------
extern "C" void kernel_launch(void* const* d_in, const int* in_sizes, int n_in,
                              void* d_out, int out_size, void* d_ws, size_t ws_size,
                              hipStream_t stream) {
  (void)in_sizes; (void)n_in; (void)out_size; (void)ws_size;
  const float* q    = (const float*)d_in[0];
  const float* k    = (const float*)d_in[1];
  const float* v    = (const float*)d_in[2];
  const float* mask = (const float*)d_in[3];
  const float* Wq   = (const float*)d_in[4];
  const float* bq   = (const float*)d_in[5];
  const float* Wk   = (const float*)d_in[6];
  const float* bk   = (const float*)d_in[7];
  const float* Wv   = (const float*)d_in[8];
  const float* bv   = (const float*)d_in[9];
  const float* Wo   = (const float*)d_in[10];
  const float* bo   = (const float*)d_in[11];
  float* out = (float*)d_out;

  char* ws = (char*)d_ws;
  size_t off = 0;
  auto carve = [&](size_t bytes) -> char* {
    char* p = ws + off;
    off += (bytes + 255) & ~(size_t)255;
    return p;
  };
  const size_t ACT_B = (size_t)NTOK * DMODEL * sizeof(__bf16);     // 8 MiB
  const size_t W_B   = (size_t)DMODEL * DMODEL * sizeof(__bf16);   // 2 MiB
  __bf16* qb    = (__bf16*)carve(ACT_B);
  __bf16* kb    = (__bf16*)carve(ACT_B);
  __bf16* vb    = (__bf16*)carve(ACT_B);
  __bf16* Wqb   = (__bf16*)carve(W_B);
  __bf16* Wkb   = (__bf16*)carve(W_B);
  __bf16* Wvb   = (__bf16*)carve(W_B);
  __bf16* Wob   = (__bf16*)carve(W_B);
  __bf16* Qp    = (__bf16*)carve(ACT_B);
  __bf16* Kp    = (__bf16*)carve(ACT_B);
  __bf16* Vp    = (__bf16*)carve(ACT_B);
  __bf16* attnb = (__bf16*)carve(ACT_B);

  // 1) f32 -> bf16 conversions
  const long nAct = (long)NTOK * DMODEL;      // 4,194,304
  const long nW   = (long)DMODEL * DMODEL;    // 1,048,576
  const int  CT   = 256;
  cvt_bf16_kernel<<<(unsigned)((nAct / 8 + CT - 1) / CT), CT, 0, stream>>>(q,  qb,  nAct);
  cvt_bf16_kernel<<<(unsigned)((nAct / 8 + CT - 1) / CT), CT, 0, stream>>>(k,  kb,  nAct);
  cvt_bf16_kernel<<<(unsigned)((nAct / 8 + CT - 1) / CT), CT, 0, stream>>>(v,  vb,  nAct);
  cvt_bf16_kernel<<<(unsigned)((nW   / 8 + CT - 1) / CT), CT, 0, stream>>>(Wq, Wqb, nW);
  cvt_bf16_kernel<<<(unsigned)((nW   / 8 + CT - 1) / CT), CT, 0, stream>>>(Wk, Wkb, nW);
  cvt_bf16_kernel<<<(unsigned)((nW   / 8 + CT - 1) / CT), CT, 0, stream>>>(Wv, Wvb, nW);
  cvt_bf16_kernel<<<(unsigned)((nW   / 8 + CT - 1) / CT), CT, 0, stream>>>(Wo, Wob, nW);

  // 2) QKV projections: y = x @ W^T + b  (bf16 out)
  dim3 gg(DMODEL / 64, NTOK / 64);  // (16, 64)
  gemm_bt_kernel<true><<<gg, 128, 0, stream>>>(qb, Wqb, bq, Qp, NTOK, DMODEL, DMODEL);
  gemm_bt_kernel<true><<<gg, 128, 0, stream>>>(kb, Wkb, bk, Kp, NTOK, DMODEL, DMODEL);
  gemm_bt_kernel<true><<<gg, 128, 0, stream>>>(vb, Wvb, bv, Vp, NTOK, DMODEL, DMODEL);

  // 3) fused flash attention -> concat layout (bf16)
  dim3 ga(S_LEN / 64, HEADS, B_DIM);  // (32, 16, 2)
  flash_attn_kernel<<<ga, 128, 0, stream>>>(Qp, Kp, Vp, mask, attnb);

  // 4) output projection: out = attn @ Wo^T + bo  (f32 out)
  gemm_bt_kernel<false><<<gg, 128, 0, stream>>>(attnb, Wob, bo, out, NTOK, DMODEL, DMODEL);
}
